// DoubleLayer_49589692399715
// MI455X (gfx1250) — matrix operand order repaired and checked by reading, not compiled, not run on hardware
//
#include <hip/hip_runtime.h>
#include <cmath>

typedef __attribute__((ext_vector_type(16))) __bf16 bf16x16;
typedef __attribute__((ext_vector_type(2)))  __bf16 bf16x2;
typedef __attribute__((ext_vector_type(8)))  float  f32x8;
typedef __attribute__((ext_vector_type(8)))  unsigned int u32x8;

#define S_IN  32
#define V_IN  16
#define S_H   64
#define V_H   32
#define S_OUT 32
#define V_OUT 16
#define NROW  80

// packed bf16 weight offsets in d_ws (ushort elements)
#define OFF_SSS1 0
#define OFF_VVS1 65536
#define OFF_SVV1 81920
#define OFF_VSV1 98304
#define OFF_SSS2 114688
#define OFF_VVS2 245760
#define OFF_SVV2 278528
#define OFF_VSV2 311296

// per-wave LDS: f32 z-staging buffer then u32 bf16-mirror buffer
#define ZS_STR  65              // staged zs rows (16 x 65)
#define ZV_STR  33              // staged zv rows (48 x 33), base +1040
#define ZBUF_U32 2624           // 16*65 + 48*33
// L1 mirrors (u32 offsets inside mirror buffer)
#define XS_PK  0                // [16][17]  packed bf16 pairs of xs
#define XS_DUP 272              // [16][33]  dup-bf16 of xs
#define XV_PK  800              // [48][9]   packed xv (row = m*3+comp)
#define XV_DUP 1232             // [48][17]
// L2 mirrors (same buffer, overwritten after fctp1)
#define S2_PK  0                // [16][33]
#define S2_DUP 528              // [16][65]
#define HV_PK  1568             // [48][17]
#define HV_DUP 2384             // [48][33]  -> ends 3968
#define MIR_U32 3968
#define W_LDS_U32 (ZBUF_U32 + MIR_U32)   // 6592 u32 per wave

__device__ __forceinline__ unsigned short f2bf_bits(float f) {
  unsigned u = __builtin_bit_cast(unsigned, f);
  u += 0x8000u;
  return (unsigned short)(u >> 16);
}
__device__ __forceinline__ float sigm(float a) { return 1.0f / (1.0f + __expf(-a)); }

__device__ __forceinline__ bf16x2 b2(unsigned u) { return __builtin_bit_cast(bf16x2, u); }
__device__ __forceinline__ unsigned ub(bf16x2 v) { return __builtin_bit_cast(unsigned, v); }

struct U4 { unsigned a, b, c, d; };
__device__ __forceinline__ U4 ld4(const unsigned* p) { return U4{p[0], p[1], p[2], p[3]}; }

__device__ __forceinline__ f32x8 wmma_bf(bf16x16 a, bf16x16 b, f32x8 c) {
  return __builtin_amdgcn_wmma_f32_16x16x32_bf16(false, a, false, b, (short)0, c, false, false);
}
__device__ __forceinline__ bf16x16 loadB(const unsigned short* p) {
  return *reinterpret_cast<const bf16x16*>(p);
}

// A builders. Pair j of the operand = elements (2j, 2j+1).
// rank-1, V-span32: pairs 0..3 = s*lo, pairs 4..7 = s*hi  (lo at u32 idx 4hi, hi at 8+4hi)
__device__ __forceinline__ bf16x16 pk_sxv(unsigned s, U4 lo, U4 hi) {
  bf16x2 sv = b2(s);
  u32x8 r;
  r[0] = ub(b2(lo.a) * sv); r[1] = ub(b2(lo.b) * sv);
  r[2] = ub(b2(lo.c) * sv); r[3] = ub(b2(lo.d) * sv);
  r[4] = ub(b2(hi.a) * sv); r[5] = ub(b2(hi.b) * sv);
  r[6] = ub(b2(hi.c) * sv); r[7] = ub(b2(hi.d) * sv);
  return __builtin_bit_cast(bf16x16, r);
}
// rank-1, V-span16 with two u-scalars: pairs 0..3 = s0*sp, pairs 4..7 = s1*sp
__device__ __forceinline__ bf16x16 pk_2sxv(unsigned s0, unsigned s1, U4 sp) {
  bf16x2 a = b2(s0), b = b2(s1);
  u32x8 r;
  r[0] = ub(b2(sp.a) * a); r[1] = ub(b2(sp.b) * a);
  r[2] = ub(b2(sp.c) * a); r[3] = ub(b2(sp.d) * a);
  r[4] = ub(b2(sp.a) * b); r[5] = ub(b2(sp.b) * b);
  r[6] = ub(b2(sp.c) * b); r[7] = ub(b2(sp.d) * b);
  return __builtin_bit_cast(bf16x16, r);
}
#define PKDOT(d0, d1, d2, x0, x1, x2) \
  ub(b2(d0) * b2(x0) + b2(d1) * b2(x1) + b2(d2) * b2(x2))
// dot3, V-span16, two u columns (a*, b*): pairs 0..3 use a-dups, 4..7 use b-dups
__device__ __forceinline__ bf16x16 pk_dot2(unsigned a0, unsigned a1, unsigned a2,
                                           unsigned b0, unsigned b1, unsigned b2v,
                                           U4 p0, U4 p1, U4 p2) {
  u32x8 r;
  r[0] = PKDOT(a0, a1, a2, p0.a, p1.a, p2.a);
  r[1] = PKDOT(a0, a1, a2, p0.b, p1.b, p2.b);
  r[2] = PKDOT(a0, a1, a2, p0.c, p1.c, p2.c);
  r[3] = PKDOT(a0, a1, a2, p0.d, p1.d, p2.d);
  r[4] = PKDOT(b0, b1, b2v, p0.a, p1.a, p2.a);
  r[5] = PKDOT(b0, b1, b2v, p0.b, p1.b, p2.b);
  r[6] = PKDOT(b0, b1, b2v, p0.c, p1.c, p2.c);
  r[7] = PKDOT(b0, b1, b2v, p0.d, p1.d, p2.d);
  return __builtin_bit_cast(bf16x16, r);
}
// dot3, V-span32, single u column: pairs 0..3 from lo-spans, 4..7 from hi-spans
__device__ __forceinline__ bf16x16 pk_dot1(unsigned d0, unsigned d1, unsigned d2,
                                           U4 l0, U4 l1, U4 l2, U4 h0, U4 h1, U4 h2) {
  u32x8 r;
  r[0] = PKDOT(d0, d1, d2, l0.a, l1.a, l2.a);
  r[1] = PKDOT(d0, d1, d2, l0.b, l1.b, l2.b);
  r[2] = PKDOT(d0, d1, d2, l0.c, l1.c, l2.c);
  r[3] = PKDOT(d0, d1, d2, l0.d, l1.d, l2.d);
  r[4] = PKDOT(d0, d1, d2, h0.a, h1.a, h2.a);
  r[5] = PKDOT(d0, d1, d2, h0.b, h1.b, h2.b);
  r[6] = PKDOT(d0, d1, d2, h0.c, h1.c, h2.c);
  r[7] = PKDOT(d0, d1, d2, h0.d, h1.d, h2.d);
  return __builtin_bit_cast(bf16x16, r);
}

// f32 weights -> bf16 in WMMA B-operand order (K = (e&7) + 8*hi + 16*(e>>3), n = t*16 + lane&15)
__global__ void prepack_kernel(const float* __restrict__ src, unsigned short* __restrict__ dst,
                               int nchunk, int ncolT, int Wout, float scale) {
  int total = ncolT * nchunk * 512;
  for (int p = blockIdx.x * blockDim.x + threadIdx.x; p < total; p += gridDim.x * blockDim.x) {
    int e = p & 15;
    int lane = (p >> 4) & 31;
    int ck = p >> 9;
    int kk = ck % nchunk;
    int t  = ck / nchunk;
    int k  = kk * 32 + (e & 7) + 8 * (lane >> 4) + 16 * (e >> 3);
    int n  = t * 16 + (lane & 15);
    dst[p] = f2bf_bits(src[k * Wout + n] * scale);
  }
}

__global__ __launch_bounds__(128) void doublelayer_kernel(
    const float* __restrict__ x, const unsigned short* __restrict__ wpk,
    const float* __restrict__ b1, const float* __restrict__ b3,
    float* __restrict__ out, int N) {
  extern __shared__ unsigned smem[];
  const int lane = threadIdx.x & 31;
  const int w    = threadIdx.x >> 5;
  const int n0   = (blockIdx.x * 4 + w) * 16;

  float*    zb = (float*)(smem + (size_t)w * W_LDS_U32);  // f32 staging, 2624 f32
  unsigned* mb = smem + (size_t)w * W_LDS_U32 + ZBUF_U32; // bf16 mirrors, 3968 u32

  // ---- load raw input tile (flat [16][80]) ----
  for (int i = lane; i < 16 * NROW; i += 32) {
    int m = i / NROW, c = i % NROW;
    int n = n0 + m;
    zb[m * NROW + c] = (n < N) ? x[(long)n * NROW + c] : 0.0f;
  }
  __syncthreads();
  // ---- gate 1: sigmoid(norm + b1); emit bf16 mirrors (packed + dup) ----
  {
    unsigned short* pks = (unsigned short*)(mb + XS_PK);
    unsigned short* pkv = (unsigned short*)(mb + XV_PK);
    for (int i = lane; i < 16 * 48; i += 32) {
      int m = i / 48, c = i % 48;
      if (c < S_IN) {
        float v = zb[m * NROW + c];
        unsigned short h = f2bf_bits(v * sigm(fabsf(v) + b1[c]));
        pks[m * 34 + c] = h;
        mb[XS_DUP + m * 33 + c] = (unsigned)h * 0x10001u;
      } else {
        int vv = c - S_IN;
        float x0 = zb[m * NROW + S_IN + vv * 3 + 0];
        float x1 = zb[m * NROW + S_IN + vv * 3 + 1];
        float x2 = zb[m * NROW + S_IN + vv * 3 + 2];
        float f = sigm(sqrtf(x0 * x0 + x1 * x1 + x2 * x2) + b1[c]);
        float g[3] = {x0 * f, x1 * f, x2 * f};
#pragma unroll
        for (int k = 0; k < 3; ++k) {
          unsigned short h = f2bf_bits(g[k]);
          pkv[(m * 3 + k) * 18 + vv] = h;
          mb[XV_DUP + (m * 3 + k) * 17 + vv] = (unsigned)h * 0x10001u;
        }
      }
    }
  }
  __syncthreads();

  const int m   = lane & 15;
  const int hi  = lane >> 4;
  const int n16 = lane & 15;
  const f32x8 zero = {0, 0, 0, 0, 0, 0, 0, 0};

  // preload layer-1 spans into VGPRs
  const unsigned* xsp = mb + XS_PK + m * 17;
  U4 xs_lo = ld4(xsp + 4 * hi);
  U4 xs_hi = ld4(xsp + 8 + 4 * hi);
  const unsigned* xsd = mb + XS_DUP + m * 33;
  U4 xv_sp[3];
  const unsigned* xvd[3];
#pragma unroll
  for (int c = 0; c < 3; ++c) {
    xv_sp[c] = ld4(mb + XV_PK + (m * 3 + c) * 9 + 4 * hi);
    xvd[c]   = mb + XV_DUP + (m * 3 + c) * 17;
  }

  float* z1s = zb;          // [16][65]
  float* z1v = zb + 1040;   // [48][33]

  // ---- fctp1: zs (16x64) ----
  {
    f32x8 a0 = zero, a1 = zero, a2 = zero, a3 = zero;
    const unsigned short* Bp = wpk + OFF_SSS1 + lane * 16;
    for (int kk = 0; kk < 32; ++kk) {           // xs (x) xs, K=1024
      bf16x16 a = pk_sxv(xsd[kk], xs_lo, xs_hi);
      a0 = wmma_bf(a, loadB(Bp + (0 * 32 + kk) * 512), a0);
      a1 = wmma_bf(a, loadB(Bp + (1 * 32 + kk) * 512), a1);
      a2 = wmma_bf(a, loadB(Bp + (2 * 32 + kk) * 512), a2);
      a3 = wmma_bf(a, loadB(Bp + (3 * 32 + kk) * 512), a3);
    }
    Bp = wpk + OFF_VVS1 + lane * 16;
    for (int kk = 0; kk < 8; ++kk) {            // xv.xv dots, K=256
      bf16x16 a = pk_dot2(xvd[0][2 * kk], xvd[1][2 * kk], xvd[2][2 * kk],
                          xvd[0][2 * kk + 1], xvd[1][2 * kk + 1], xvd[2][2 * kk + 1],
                          xv_sp[0], xv_sp[1], xv_sp[2]);
      a0 = wmma_bf(a, loadB(Bp + (0 * 8 + kk) * 512), a0);
      a1 = wmma_bf(a, loadB(Bp + (1 * 8 + kk) * 512), a1);
      a2 = wmma_bf(a, loadB(Bp + (2 * 8 + kk) * 512), a2);
      a3 = wmma_bf(a, loadB(Bp + (3 * 8 + kk) * 512), a3);
    }
#pragma unroll
    for (int r = 0; r < 8; ++r) {
      int M = r + 8 * hi;
      z1s[M * ZS_STR +  0 + n16] = a0[r];
      z1s[M * ZS_STR + 16 + n16] = a1[r];
      z1s[M * ZS_STR + 32 + n16] = a2[r];
      z1s[M * ZS_STR + 48 + n16] = a3[r];
    }
  }
  // ---- fctp1: zv (16x32 x 3 comps), B chunks shared across comps ----
  {
    f32x8 c00 = zero, c01 = zero, c10 = zero, c11 = zero, c20 = zero, c21 = zero;
    const unsigned short* Bp = wpk + OFF_SVV1 + lane * 16;
    for (int kk = 0; kk < 16; ++kk) {           // xs (x) xv_k, K=512 (u*16+v)
      unsigned s0 = xsd[2 * kk], s1 = xsd[2 * kk + 1];
      bf16x16 B0 = loadB(Bp + (0 * 16 + kk) * 512);
      bf16x16 B1 = loadB(Bp + (1 * 16 + kk) * 512);
      bf16x16 aa0 = pk_2sxv(s0, s1, xv_sp[0]);
      bf16x16 aa1 = pk_2sxv(s0, s1, xv_sp[1]);
      bf16x16 aa2 = pk_2sxv(s0, s1, xv_sp[2]);
      c00 = wmma_bf(aa0, B0, c00); c01 = wmma_bf(aa0, B1, c01);
      c10 = wmma_bf(aa1, B0, c10); c11 = wmma_bf(aa1, B1, c11);
      c20 = wmma_bf(aa2, B0, c20); c21 = wmma_bf(aa2, B1, c21);
    }
    Bp = wpk + OFF_VSV1 + lane * 16;
    for (int kk = 0; kk < 16; ++kk) {           // xv_k (x) xs, K=512 (u*32+v)
      bf16x16 B0 = loadB(Bp + (0 * 16 + kk) * 512);
      bf16x16 B1 = loadB(Bp + (1 * 16 + kk) * 512);
      bf16x16 aa0 = pk_sxv(xvd[0][kk], xs_lo, xs_hi);
      bf16x16 aa1 = pk_sxv(xvd[1][kk], xs_lo, xs_hi);
      bf16x16 aa2 = pk_sxv(xvd[2][kk], xs_lo, xs_hi);
      c00 = wmma_bf(aa0, B0, c00); c01 = wmma_bf(aa0, B1, c01);
      c10 = wmma_bf(aa1, B0, c10); c11 = wmma_bf(aa1, B1, c11);
      c20 = wmma_bf(aa2, B0, c20); c21 = wmma_bf(aa2, B1, c21);
    }
#pragma unroll
    for (int r = 0; r < 8; ++r) {
      int M = r + 8 * hi;
      z1v[(M * 3 + 0) * ZV_STR +  0 + n16] = c00[r];
      z1v[(M * 3 + 0) * ZV_STR + 16 + n16] = c01[r];
      z1v[(M * 3 + 1) * ZV_STR +  0 + n16] = c10[r];
      z1v[(M * 3 + 1) * ZV_STR + 16 + n16] = c11[r];
      z1v[(M * 3 + 2) * ZV_STR +  0 + n16] = c20[r];
      z1v[(M * 3 + 2) * ZV_STR + 16 + n16] = c21[r];
    }
  }
  __syncthreads();
  // ---- norm_act layer 2 (normalize): emit L2 bf16 mirrors (overwrite L1 mirrors) ----
  {
    unsigned short* pks = (unsigned short*)(mb + S2_PK);
    unsigned short* pkh = (unsigned short*)(mb + HV_PK);
    for (int i = lane; i < 16 * 96; i += 32) {
      int mm = i / 96, c = i % 96;
      if (c < S_H) {
        float v = zb[mm * ZS_STR + c];
        float nm = fabsf(v);
        unsigned short h = f2bf_bits(v * (sigm(nm) / nm));
        pks[mm * 66 + c] = h;
        mb[S2_DUP + mm * 65 + c] = (unsigned)h * 0x10001u;
      } else {
        int vv = c - S_H;
        float x0 = z1v[(mm * 3 + 0) * ZV_STR + vv];
        float x1 = z1v[(mm * 3 + 1) * ZV_STR + vv];
        float x2 = z1v[(mm * 3 + 2) * ZV_STR + vv];
        float nm = sqrtf(x0 * x0 + x1 * x1 + x2 * x2);
        float f = sigm(nm) / nm;
        float g[3] = {x0 * f, x1 * f, x2 * f};
#pragma unroll
        for (int k = 0; k < 3; ++k) {
          unsigned short h = f2bf_bits(g[k]);
          pkh[(mm * 3 + k) * 34 + vv] = h;
          mb[HV_DUP + (mm * 3 + k) * 33 + vv] = (unsigned)h * 0x10001u;
        }
      }
    }
  }
  __syncthreads();

  // preload layer-2 spans
  const unsigned* s2p = mb + S2_PK + m * 33;
  U4 s2_ll = ld4(s2p + 4 * hi);        // v in [ 0..31] low pairs
  U4 s2_lh = ld4(s2p + 8 + 4 * hi);    // v in [ 0..31] high pairs
  U4 s2_hl = ld4(s2p + 16 + 4 * hi);   // v in [32..63] low pairs
  U4 s2_hh = ld4(s2p + 24 + 4 * hi);   // v in [32..63] high pairs
  const unsigned* s2d = mb + S2_DUP + m * 65;
  U4 hv_lo[3], hv_hi[3];
  const unsigned* hvd[3];
#pragma unroll
  for (int c = 0; c < 3; ++c) {
    const unsigned* p = mb + HV_PK + (m * 3 + c) * 17;
    hv_lo[c] = ld4(p + 4 * hi);
    hv_hi[c] = ld4(p + 8 + 4 * hi);
    hvd[c]   = mb + HV_DUP + (m * 3 + c) * 33;
  }

  // ---- fctp2: os (16x32); one scalar feeds 4 WMMAs ----
  {
    f32x8 a0 = zero, a1 = zero;
    const unsigned short* Bp = wpk + OFF_SSS2 + lane * 16;
    for (int u = 0; u < 64; ++u) {              // hs (x) hs, K=4096 (u*64+v)
      unsigned sd = s2d[u];
      bf16x16 aL = pk_sxv(sd, s2_ll, s2_lh);
      a0 = wmma_bf(aL, loadB(Bp + (0 * 128 + 2 * u) * 512), a0);
      a1 = wmma_bf(aL, loadB(Bp + (1 * 128 + 2 * u) * 512), a1);
      bf16x16 aH = pk_sxv(sd, s2_hl, s2_hh);
      a0 = wmma_bf(aH, loadB(Bp + (0 * 128 + 2 * u + 1) * 512), a0);
      a1 = wmma_bf(aH, loadB(Bp + (1 * 128 + 2 * u + 1) * 512), a1);
    }
    Bp = wpk + OFF_VVS2 + lane * 16;
    for (int kk = 0; kk < 32; ++kk) {           // hv.hv dots, K=1024 (u*32+v)
      bf16x16 a = pk_dot1(hvd[0][kk], hvd[1][kk], hvd[2][kk],
                          hv_lo[0], hv_lo[1], hv_lo[2],
                          hv_hi[0], hv_hi[1], hv_hi[2]);
      a0 = wmma_bf(a, loadB(Bp + (0 * 32 + kk) * 512), a0);
      a1 = wmma_bf(a, loadB(Bp + (1 * 32 + kk) * 512), a1);
    }
#pragma unroll
    for (int r = 0; r < 8; ++r) {
      int M = r + 8 * hi;
      zb[M * NROW +  0 + n16] = a0[r];
      zb[M * NROW + 16 + n16] = a1[r];
    }
  }
  // ---- fctp2: ov (16x16 x 3 comps), B shared across comps ----
  {
    f32x8 v0 = zero, v1 = zero, v2 = zero;
    const unsigned short* Bp = wpk + OFF_SVV2 + lane * 16;
    for (int kk = 0; kk < 64; ++kk) {           // hs (x) hv_k, K=2048 (u*32+v)
      unsigned sd = s2d[kk];
      bf16x16 B = loadB(Bp + kk * 512);
      v0 = wmma_bf(pk_sxv(sd, hv_lo[0], hv_hi[0]), B, v0);
      v1 = wmma_bf(pk_sxv(sd, hv_lo[1], hv_hi[1]), B, v1);
      v2 = wmma_bf(pk_sxv(sd, hv_lo[2], hv_hi[2]), B, v2);
    }
    Bp = wpk + OFF_VSV2 + lane * 16;
    for (int u = 0; u < 32; ++u) {              // hv_k (x) hs, K=2048 (u*64+v)
      unsigned h0 = hvd[0][u], h1 = hvd[1][u], h2 = hvd[2][u];
      bf16x16 BL = loadB(Bp + (2 * u) * 512);
      v0 = wmma_bf(pk_sxv(h0, s2_ll, s2_lh), BL, v0);
      v1 = wmma_bf(pk_sxv(h1, s2_ll, s2_lh), BL, v1);
      v2 = wmma_bf(pk_sxv(h2, s2_ll, s2_lh), BL, v2);
      bf16x16 BH = loadB(Bp + (2 * u + 1) * 512);
      v0 = wmma_bf(pk_sxv(h0, s2_hl, s2_hh), BH, v0);
      v1 = wmma_bf(pk_sxv(h1, s2_hl, s2_hh), BH, v1);
      v2 = wmma_bf(pk_sxv(h2, s2_hl, s2_hh), BH, v2);
    }
#pragma unroll
    for (int r = 0; r < 8; ++r) {
      int M = r + 8 * hi;
      zb[M * NROW + S_OUT + n16 * 3 + 0] = v0[r];
      zb[M * NROW + S_OUT + n16 * 3 + 1] = v1[r];
      zb[M * NROW + S_OUT + n16 * 3 + 2] = v2[r];
    }
  }
  __syncthreads();
  // ---- final gate (bias b3) + store ----
  for (int i = lane; i < 16 * 48; i += 32) {
    int mm = i / 48, c = i % 48;
    int n = n0 + mm;
    const float* row = zb + mm * NROW;
    if (c < S_OUT) {
      float v = row[c];
      if (n < N) out[(long)n * NROW + c] = v * sigm(fabsf(v) + b3[c]);
    } else {
      int vv = c - S_OUT;
      float x0 = row[S_OUT + vv * 3 + 0];
      float x1 = row[S_OUT + vv * 3 + 1];
      float x2 = row[S_OUT + vv * 3 + 2];
      float f = sigm(sqrtf(x0 * x0 + x1 * x1 + x2 * x2) + b3[c]);
      if (n < N) {
        out[(long)n * NROW + S_OUT + vv * 3 + 0] = x0 * f;
        out[(long)n * NROW + S_OUT + vv * 3 + 1] = x1 * f;
        out[(long)n * NROW + S_OUT + vv * 3 + 2] = x2 * f;
      }
    }
  }
}

extern "C" void kernel_launch(void* const* d_in, const int* in_sizes, int n_in,
                              void* d_out, int out_size, void* d_ws, size_t ws_size,
                              hipStream_t stream) {
  const float* x     = (const float*)d_in[0];
  const float* Wsss1 = (const float*)d_in[1];
  const float* Wvvs1 = (const float*)d_in[2];
  const float* Wsvv1 = (const float*)d_in[3];
  const float* Wvsv1 = (const float*)d_in[4];
  const float* Wsss2 = (const float*)d_in[5];
  const float* Wvvs2 = (const float*)d_in[6];
  const float* Wsvv2 = (const float*)d_in[7];
  const float* Wvsv2 = (const float*)d_in[8];
  const float* b1    = (const float*)d_in[9];
  const float* b3    = (const float*)d_in[10];
  float* out = (float*)d_out;
  unsigned short* wpk = (unsigned short*)d_ws;
  int N = in_sizes[0] / NROW;

  const float INV3 = (float)(1.0 / sqrt(3.0));
  const float pw1s = (float)sqrt(1.0 / (double)(S_IN * S_IN + V_IN * V_IN));
  const float pw1v = (float)sqrt(3.0 / (double)(S_IN * V_IN + V_IN * S_IN));
  const float pw2s = (float)sqrt(1.0 / (double)(S_H * S_H + V_H * V_H));
  const float pw2v = (float)sqrt(3.0 / (double)(S_H * V_H + V_H * S_H));

  auto pp = [&](const float* src, int off, int nchunk, int ncolT, int Wout, float scale) {
    int total = ncolT * nchunk * 512;
    int blocks = (total + 255) / 256;
    if (blocks > 1024) blocks = 1024;
    prepack_kernel<<<blocks, 256, 0, stream>>>(src, wpk + off, nchunk, ncolT, Wout, scale);
  };
  pp(Wsss1, OFF_SSS1,  32, 4, 64, pw1s);
  pp(Wvvs1, OFF_VVS1,   8, 4, 64, pw1s * INV3);
  pp(Wsvv1, OFF_SVV1,  16, 2, 32, pw1v * INV3);
  pp(Wvsv1, OFF_VSV1,  16, 2, 32, pw1v * INV3);
  pp(Wsss2, OFF_SSS2, 128, 2, 32, pw2s);
  pp(Wvvs2, OFF_VVS2,  32, 2, 32, pw2s * INV3);
  pp(Wsvv2, OFF_SVV2,  64, 1, 16, pw2v * INV3);
  pp(Wvsv2, OFF_VSV2,  64, 1, 16, pw2v * INV3);

  int tiles  = (N + 15) / 16;
  int blocks = (tiles + 3) / 4;
  size_t lds = (size_t)4 * W_LDS_U32 * 4;   // 105472 bytes
  doublelayer_kernel<<<blocks, 128, lds, stream>>>(x, wpk, b1, b3, out, N);
}
